// CausalSelfAttention_70076686402323
// MI455X (gfx1250) — matrix operand order repaired and checked by reading, not compiled, not run on hardware
//
#include <hip/hip_runtime.h>
#include <stdint.h>

#define BSZ   4
#define TSEQ  2048
#define CDIM  1024
#define NHEAD 16
#define HSZ   64
#define C3    3072
#define MTOT  (BSZ * TSEQ)   // 8192

typedef __attribute__((ext_vector_type(16))) __bf16 v16bf;
typedef __attribute__((ext_vector_type(2)))  __bf16 v2bf;
typedef __attribute__((ext_vector_type(8)))  float  v8f;
typedef __attribute__((ext_vector_type(4)))  int    v4i;

// ---------------- f32 -> bf16 (RNE) ----------------
__device__ __forceinline__ unsigned short f2bf(float f) {
  union { float f; uint32_t u; } c{f};
  uint32_t u = c.u;
  u += 0x7FFFu + ((u >> 16) & 1u);
  return (unsigned short)(u >> 16);
}
__device__ __forceinline__ uint32_t f2bf2(float lo, float hi) {
#if __has_builtin(__builtin_amdgcn_cvt_pk_bf16_f32)
  union { v2bf v; uint32_t u; } c;
  c.v = __builtin_amdgcn_cvt_pk_bf16_f32(lo, hi);
  return c.u;
#else
  return (uint32_t)f2bf(lo) | ((uint32_t)f2bf(hi) << 16);
#endif
}

// ---------------- async global -> LDS (CDNA5) ----------------
#if __has_builtin(__builtin_amdgcn_global_load_async_to_lds_b128)
#define HAVE_ASYNC_LDS 1
#else
#define HAVE_ASYNC_LDS 0
#endif

typedef __attribute__((address_space(1))) v4i* gptr_b128;
typedef __attribute__((address_space(3))) v4i* lptr_b128;

__device__ __forceinline__ void async_copy16(const unsigned short* g,
                                             unsigned short* l) {
#if HAVE_ASYNC_LDS
  __builtin_amdgcn_global_load_async_to_lds_b128(
      (gptr_b128)(void*)g, (lptr_b128)(void*)l, 0, 0);
#else
  *reinterpret_cast<uint4*>(l) = *reinterpret_cast<const uint4*>(g);
#endif
}
__device__ __forceinline__ void async_wait() {
#if HAVE_ASYNC_LDS
#if __has_builtin(__builtin_amdgcn_s_wait_asynccnt)
  __builtin_amdgcn_s_wait_asynccnt(0);
#else
  asm volatile("s_wait_asynccnt 0" ::: "memory");
#endif
#endif
}

// ---------------- WMMA fragment loads ----------------
union FragU { v16bf v; uint4 q[2]; };

// A-matrix 16x32 bf16: lane half -> contiguous K 0..15 / 16..31
__device__ __forceinline__ v16bf ldsA(const unsigned short* rowPtr, int kw, int half) {
  FragU f;
  const int kb = kw + half * 16;
  f.q[0] = *reinterpret_cast<const uint4*>(rowPtr + kb);
  f.q[1] = *reinterpret_cast<const uint4*>(rowPtr + kb + 8);
  return f.v;
}
// B-matrix 32x16 bf16: lane half -> contiguous K 0..15 / 16..31
__device__ __forceinline__ v16bf ldsB(const unsigned short* rowPtr, int kw, int half) {
  FragU f;
  const int kb = kw + half * 16;
  f.q[0] = *reinterpret_cast<const uint4*>(rowPtr + kb);
  f.q[1] = *reinterpret_cast<const uint4*>(rowPtr + kb + 8);
  return f.v;
}
__device__ __forceinline__ v8f wmma_bf16(v16bf a, v16bf b, v8f c) {
  return __builtin_amdgcn_wmma_f32_16x16x32_bf16(false, a, false, b, (short)0, c,
                                                 false, false);
}

// ---------------------------------------------------------------------------
// Kernel 0: bulk f32 -> bf16 conversion (one pass; removes per-tile re-convert)
// ---------------------------------------------------------------------------
__global__ __launch_bounds__(256) void cvt_kernel(const float* __restrict__ src,
                                                  unsigned short* __restrict__ dst,
                                                  int n8) {
  const int i = blockIdx.x * 256 + threadIdx.x;
  if (i >= n8) return;
  const float4* s = reinterpret_cast<const float4*>(src) + (size_t)i * 2;
  float4 a0 = s[0], a1 = s[1];
  uint4 p;
  p.x = f2bf2(a0.x, a0.y); p.y = f2bf2(a0.z, a0.w);
  p.z = f2bf2(a1.x, a1.y); p.w = f2bf2(a1.z, a1.w);
  reinterpret_cast<uint4*>(dst)[i] = p;
}

// ---------------------------------------------------------------------------
// Kernel 1: qkv = x @ Wqkv^T + bqkv  -> q*scale, k, v in [B,NH,T,HS] bf16
// block 256 thr (8 waves); tile M=128 x N=64; K step 32; bf16 async staging
// ---------------------------------------------------------------------------
__global__ __launch_bounds__(256) void qkv_kernel(
    const unsigned short* __restrict__ xw, const unsigned short* __restrict__ Ww,
    const float* __restrict__ bias,
    unsigned short* __restrict__ qw, unsigned short* __restrict__ kw_,
    unsigned short* __restrict__ vw)
{
  __shared__ __align__(16) unsigned short xs[128 * 40];
  __shared__ __align__(16) unsigned short wsb[64 * 40];
  const int tid  = threadIdx.x;
  const int wave = tid >> 5, lane = tid & 31;
  const int idx  = lane & 15, half = lane >> 4;
  const int nBase = blockIdx.x * 64;
  const int mBase = blockIdx.y * 128;

  v8f acc[4] = {{}, {}, {}, {}};

  for (int kb = 0; kb < CDIM; kb += 32) {
    { // stage A: 128x32 bf16, 2 x b128 per thread (async)
#pragma unroll
      for (int i = 0; i < 2; ++i) {
        const int c = tid + 256 * i;           // chunk id, 4 chunks per row
        const int row = c >> 2, part = (c & 3) * 8;
        async_copy16(xw + (size_t)(mBase + row) * CDIM + kb + part,
                     &xs[row * 40 + part]);
      }
    }
    { // stage B: 64x32 bf16, 1 x b128 per thread (async)
      const int row = tid >> 2, part = (tid & 3) * 8;
      async_copy16(Ww + (size_t)(nBase + row) * CDIM + kb + part,
                   &wsb[row * 40 + part]);
    }
    async_wait();
    __syncthreads();
    v16bf a = ldsA(&xs[(wave * 16 + idx) * 40], 0, half);
#pragma unroll
    for (int nt = 0; nt < 4; ++nt) {
      v16bf b = ldsB(&wsb[(nt * 16 + idx) * 40], 0, half);
      acc[nt] = wmma_bf16(a, b, acc[nt]);
    }
    __syncthreads();
  }

  // epilogue: + bias, q scaled by 1/sqrt(HS), scatter to [B,NH,T,HS]
#pragma unroll
  for (int nt = 0; nt < 4; ++nt) {
    const int n = nBase + nt * 16 + idx;
    const int sel = n >> 10, c = n & 1023;
    const int h = c >> 6, d = c & 63;
    const float bv = bias[n];
    unsigned short* dstBase = (sel == 0) ? qw : (sel == 1) ? kw_ : vw;
    const float sc = (sel == 0) ? 0.125f : 1.0f;  // 1/sqrt(64)
#pragma unroll
    for (int r = 0; r < 8; ++r) {
      const int m = mBase + wave * 16 + half * 8 + r;
      const int b = m >> 11, t = m & 2047;
      const float v = (acc[nt][r] + bv) * sc;
      dstBase[(((size_t)(b * NHEAD + h)) * TSEQ + t) * HSZ + d] = f2bf(v);
    }
  }
}

// ---------------------------------------------------------------------------
// Kernel 2: flash-attention tile. block = (q-tile of 128) x (b,h).
// ---------------------------------------------------------------------------
__global__ __launch_bounds__(256) void attn_kernel(
    const unsigned short* __restrict__ qw, const unsigned short* __restrict__ kw_,
    const unsigned short* __restrict__ vw, unsigned short* __restrict__ yw)
{
  __shared__ __align__(16) unsigned short qs[128 * 72];
  __shared__ __align__(16) unsigned short ks[64 * 72];
  __shared__ __align__(16) unsigned short vts[64 * 72];   // transposed: [d][j]
  __shared__ __align__(16) unsigned short ps[8][16 * 72]; // per-wave P tile

  const int tid  = threadIdx.x;
  const int wave = tid >> 5, lane = tid & 31;
  const int idx  = lane & 15, half = lane >> 4;
  const int qBase = blockIdx.x * 128;
  const int bh = blockIdx.y;
  const size_t headOff = (size_t)bh * TSEQ * HSZ;

  { // stage Q tile once: 128x64 bf16, 4 x b128 per thread (async)
#pragma unroll
    for (int i = 0; i < 4; ++i) {
      const int c = tid + 256 * i;             // 8 chunks per 64-elem row
      const int row = c >> 3, part = (c & 7) * 8;
      async_copy16(qw + headOff + (size_t)(qBase + row) * HSZ + part,
                   &qs[row * 72 + part]);
    }
  }

  v8f o[4] = {{}, {}, {}, {}};
  float mrun[8], lrun[8];
#pragma unroll
  for (int r = 0; r < 8; ++r) { mrun[r] = -1e30f; lrun[r] = 0.0f; }

  const int iRow0 = qBase + wave * 16 + half * 8;  // +r -> query index

  for (int jb = 0; jb <= qBase + 127; jb += 64) {  // causal: skip dead tiles
    __syncthreads();
    { // stage K tile 64x64 (async)
#pragma unroll
      for (int i = 0; i < 2; ++i) {
        const int c = tid + 256 * i;
        const int row = c >> 3, part = (c & 7) * 8;
        async_copy16(kw_ + headOff + (size_t)(jb + row) * HSZ + part,
                     &ks[row * 72 + part]);
      }
    }
    { // stage V tile transposed -> vts[d][j] (sync scatter)
      const int row = tid >> 2, part = (tid & 3) * 16;
      union { uint4 q[2]; unsigned short s[16]; } vv;
      const uint4* src = reinterpret_cast<const uint4*>(
          vw + headOff + (size_t)(jb + row) * HSZ + part);
      vv.q[0] = src[0]; vv.q[1] = src[1];
#pragma unroll
      for (int e = 0; e < 16; ++e)
        vts[(part + e) * 72 + row] = vv.s[e];
    }
    async_wait();
    __syncthreads();

    // S = Q K^T  (16x64 per wave, K-dim = HS = 64)
    v8f s[4] = {{}, {}, {}, {}};
#pragma unroll
    for (int kk = 0; kk < 64; kk += 32) {
      v16bf a = ldsA(&qs[(wave * 16 + idx) * 72], kk, half);
#pragma unroll
      for (int nt = 0; nt < 4; ++nt) {
        v16bf b = ldsB(&ks[(nt * 16 + idx) * 72], kk, half);
        s[nt] = wmma_bf16(a, b, s[nt]);
      }
    }

    // causal mask + row max (rows live within 16-lane halves in C-layout)
    float rmax[8], rsum[8];
#pragma unroll
    for (int r = 0; r < 8; ++r) rmax[r] = -1e30f;
#pragma unroll
    for (int nt = 0; nt < 4; ++nt) {
      const int j = jb + nt * 16 + idx;
#pragma unroll
      for (int r = 0; r < 8; ++r) {
        const int i = iRow0 + r;
        const float val = (j > i) ? -1e30f : s[nt][r];
        s[nt][r] = val;
        rmax[r] = fmaxf(rmax[r], val);
      }
    }
#pragma unroll
    for (int r = 0; r < 8; ++r)
#pragma unroll
      for (int msk = 1; msk < 16; msk <<= 1)
        rmax[r] = fmaxf(rmax[r], __shfl_xor(rmax[r], msk, 32));

    float rescale[8];
#pragma unroll
    for (int r = 0; r < 8; ++r) {
      const float mnew = fmaxf(mrun[r], rmax[r]);
      rescale[r] = __expf(mrun[r] - mnew);
      mrun[r] = mnew;
      rsum[r] = 0.0f;
    }
#pragma unroll
    for (int nt = 0; nt < 4; ++nt)
#pragma unroll
      for (int r = 0; r < 8; ++r) {
        const float p = __expf(s[nt][r] - mrun[r]);
        s[nt][r] = p;
        rsum[r] += p;
      }
#pragma unroll
    for (int r = 0; r < 8; ++r) {
#pragma unroll
      for (int msk = 1; msk < 16; msk <<= 1)
        rsum[r] += __shfl_xor(rsum[r], msk, 32);
      lrun[r] = lrun[r] * rescale[r] + rsum[r];
    }
#pragma unroll
    for (int nt = 0; nt < 4; ++nt)
#pragma unroll
      for (int r = 0; r < 8; ++r)
        o[nt][r] *= rescale[r];

    // P: C-layout -> A-layout via this wave's private LDS region
    unsigned short* pw = &ps[wave][0];
#pragma unroll
    for (int nt = 0; nt < 4; ++nt)
#pragma unroll
      for (int r = 0; r < 8; ++r)
        pw[(half * 8 + r) * 72 + nt * 16 + idx] = f2bf(s[nt][r]);
    asm volatile("s_wait_dscnt 0" ::: "memory");  // wave-local DS RAW fence

    // O += P @ V   (K-dim = 64 keys)
#pragma unroll
    for (int kk = 0; kk < 64; kk += 32) {
      v16bf a = ldsA(&pw[idx * 72], kk, half);
#pragma unroll
      for (int nt = 0; nt < 4; ++nt) {
        v16bf b = ldsB(&vts[(nt * 16 + idx) * 72], kk, half);
        o[nt] = wmma_bf16(a, b, o[nt]);
      }
    }
  }

  // finalize: O /= l, store y[B,T,C] bf16
  const int b = bh / NHEAD, h = bh % NHEAD;
#pragma unroll
  for (int r = 0; r < 8; ++r) {
    const float inv = 1.0f / lrun[r];
    const int t = iRow0 + r;
    const size_t rowOff = ((size_t)(b * TSEQ + t)) * CDIM + h * HSZ;
#pragma unroll
    for (int nt = 0; nt < 4; ++nt)
      yw[rowOff + nt * 16 + idx] = f2bf(o[nt][r] * inv);
  }
}

// ---------------------------------------------------------------------------
// Kernel 3: out = y @ Wproj^T + bproj   (y bf16, Wproj pre-converted bf16)
// ---------------------------------------------------------------------------
__global__ __launch_bounds__(256) void proj_kernel(
    const unsigned short* __restrict__ yw, const unsigned short* __restrict__ Ww,
    const float* __restrict__ bias, float* __restrict__ out)
{
  __shared__ __align__(16) unsigned short xs[128 * 40];
  __shared__ __align__(16) unsigned short wsb[64 * 40];
  const int tid  = threadIdx.x;
  const int wave = tid >> 5, lane = tid & 31;
  const int idx  = lane & 15, half = lane >> 4;
  const int nBase = blockIdx.x * 64;
  const int mBase = blockIdx.y * 128;

  v8f acc[4] = {{}, {}, {}, {}};

  for (int kb = 0; kb < CDIM; kb += 32) {
#pragma unroll
    for (int i = 0; i < 2; ++i) {             // stage A (async)
      const int c = tid + 256 * i;
      const int row = c >> 2, part = (c & 3) * 8;
      async_copy16(yw + (size_t)(mBase + row) * CDIM + kb + part,
                   &xs[row * 40 + part]);
    }
    {                                          // stage B (async)
      const int row = tid >> 2, part = (tid & 3) * 8;
      async_copy16(Ww + (size_t)(nBase + row) * CDIM + kb + part,
                   &wsb[row * 40 + part]);
    }
    async_wait();
    __syncthreads();
    v16bf a = ldsA(&xs[(wave * 16 + idx) * 40], 0, half);
#pragma unroll
    for (int nt = 0; nt < 4; ++nt) {
      v16bf b = ldsB(&wsb[(nt * 16 + idx) * 40], 0, half);
      acc[nt] = wmma_bf16(a, b, acc[nt]);
    }
    __syncthreads();
  }

#pragma unroll
  for (int nt = 0; nt < 4; ++nt) {
    const int n = nBase + nt * 16 + idx;
    const float bv = bias[n];
#pragma unroll
    for (int r = 0; r < 8; ++r) {
      const int m = mBase + wave * 16 + half * 8 + r;
      out[(size_t)m * CDIM + n] = acc[nt][r] + bv;
    }
  }
}

// ---------------------------------------------------------------------------
extern "C" void kernel_launch(void* const* d_in, const int* in_sizes, int n_in,
                              void* d_out, int out_size, void* d_ws, size_t ws_size,
                              hipStream_t stream) {
  const float* x     = (const float*)d_in[0];
  const float* Wqkv  = (const float*)d_in[1];
  const float* bqkv  = (const float*)d_in[2];
  const float* Wproj = (const float*)d_in[3];
  const float* bproj = (const float*)d_in[4];
  float* out = (float*)d_out;

  char* ws = (char*)d_ws;
  const size_t MB = 1024 * 1024;
  unsigned short* qw  = (unsigned short*)(ws);            // 16 MB
  unsigned short* kw  = (unsigned short*)(ws + 16 * MB);  // 16 MB
  unsigned short* vw  = (unsigned short*)(ws + 32 * MB);  // 16 MB
  unsigned short* yw  = (unsigned short*)(ws + 48 * MB);  // 16 MB
  unsigned short* xw  = (unsigned short*)(ws + 64 * MB);  // 16 MB
  unsigned short* Wqw = (unsigned short*)(ws + 80 * MB);  //  6 MB
  unsigned short* Wpw = (unsigned short*)(ws + 86 * MB);  //  2 MB

  dim3 blk(256);
  const int nx8 = MTOT * CDIM / 8, nwq8 = C3 * CDIM / 8, nwp8 = CDIM * CDIM / 8;
  cvt_kernel<<<dim3((nx8  + 255) / 256), blk, 0, stream>>>(x,     xw,  nx8);
  cvt_kernel<<<dim3((nwq8 + 255) / 256), blk, 0, stream>>>(Wqkv,  Wqw, nwq8);
  cvt_kernel<<<dim3((nwp8 + 255) / 256), blk, 0, stream>>>(Wproj, Wpw, nwp8);

  qkv_kernel<<<dim3(C3 / 64, MTOT / 128), blk, 0, stream>>>(xw, Wqw, bqkv, qw, kw, vw);
  attn_kernel<<<dim3(TSEQ / 128, BSZ * NHEAD), blk, 0, stream>>>(qw, kw, vw, yw);
  proj_kernel<<<dim3(CDIM / 64, MTOT / 128), blk, 0, stream>>>(yw, Wpw, bproj, out);
}
